// RQVAE_5720896438811
// MI455X (gfx1250) — matrix operand order repaired and checked by reading, not compiled, not run on hardware
//
#include <hip/hip_runtime.h>
#include <math.h>

// ---------------------------------------------------------------------------
// CDNA5 (gfx1250) wave32 WMMA types
// ---------------------------------------------------------------------------
typedef _Float16 v16h __attribute__((ext_vector_type(16)));
typedef float    v8f  __attribute__((ext_vector_type(8)));

#define WMMA_F16(a, b, c) \
  __builtin_amdgcn_wmma_f32_16x16x32_f16(false, (a), false, (b), (short)0, (c), false, false)

// ---------------------------------------------------------------------------
// Generic implicit-GEMM convolution with fused epilogue.
//   out[b, n, oy, ox] = relu?( bias[n] + res? + sum_k A[m,k]*W[n,k] )
//   M = B*OH*OW, N = OC, K = IC*KH*KW
// Supports stride, zero padding, lhs_dilation (transposed conv), input ReLU.
// Block: 256 threads = 8 waves. Block tile: 128(M) x 64(N), K in steps of 32.
// Each wave owns 16 M-rows and all 64 N-cols (4 x v8f accumulators,
// 4 WMMAs per K-step reusing one A fragment -> 4x A-traffic reduction).
// A fragment (16x32 f16): lane<16 -> K {0..7,16..23}, lane>=16 -> K {8..15,24..31}, m = lane&15
// B fragment (32x16 f16): n = lane&15, same K per lane and element as A
// C/D layout: VGPR v, lanes0-15 -> M=v, N=lane; lanes16-31 -> M=v+8, N=lane-16
// ---------------------------------------------------------------------------
__global__ __launch_bounds__(256) void conv_wmma(
    const float* __restrict__ in, const float* __restrict__ w,
    const float* __restrict__ bias, const float* __restrict__ res,
    float* __restrict__ out,
    int B, int IC, int IH, int IW, int OC, int OH, int OW,
    int KH, int KW, int stride, int pad, int ldil,
    int relu_in, int relu_out)
{
  __shared__ _Float16 Alds[128 * 32];   // [m_local][k]   8 KB
  __shared__ _Float16 Blds[32 * 64];    // [k][n]         4 KB

  const int t    = threadIdx.x;
  const int wave = t >> 5;
  const int lane = t & 31;
  const int Mtot = B * OH * OW;
  const int KHW  = KH * KW;
  const int Ktot = IC * KHW;
  const int m0   = blockIdx.x * 128;
  const int n0   = blockIdx.y * 64;

  // Per-thread A staging coordinates (row fixed for the whole K loop)
  const int arow = t >> 1;          // 0..127
  const int akc  = (t & 1) << 4;    // 0 or 16
  const int am   = m0 + arow;
  const bool amv = (am < Mtot);
  int ab = 0, aoy = 0, aox = 0;
  if (amv) { ab = am / (OH * OW); int r = am % (OH * OW); aoy = r / OW; aox = r % OW; }

  v8f acc[4] = {};
  const int ksteps = (Ktot + 31) >> 5;

  for (int ks = 0; ks < ksteps; ++ks) {
    const int k0 = ks << 5;

    // ---- stage A tile (im2col gather, fp32 -> f16, optional input ReLU) ----
    #pragma unroll 4
    for (int j = 0; j < 16; ++j) {
      const int k = k0 + akc + j;
      float v = 0.f;
      if (amv && k < Ktot) {
        const int ic = k / KHW;
        const int r2 = k - ic * KHW;
        const int ky = r2 / KW;
        const int kx = r2 - ky * KW;
        const int vy = aoy * stride - pad + ky;
        const int vx = aox * stride - pad + kx;
        if (ldil == 1) {
          if (vy >= 0 && vy < IH && vx >= 0 && vx < IW)
            v = in[((ab * IC + ic) * IH + vy) * IW + vx];
        } else {
          if (vy >= 0 && vx >= 0 && (vy % ldil) == 0 && (vx % ldil) == 0) {
            const int iy = vy / ldil, ix = vx / ldil;
            if (iy < IH && ix < IW)
              v = in[((ab * IC + ic) * IH + iy) * IW + ix];
          }
        }
        if (relu_in) v = v > 0.f ? v : 0.f;
      }
      Alds[arow * 32 + akc + j] = (_Float16)v;
    }

    // ---- stage B tile: weights [N][K] row-major -> LDS [k][n], n in [0,64) ----
    #pragma unroll
    for (int e = t; e < 2048; e += 256) {
      const int n  = e & 63, kk = e >> 6;
      const int k  = k0 + kk;
      const int gn = n0 + n;
      float v = 0.f;
      if (gn < OC && k < Ktot) v = w[gn * Ktot + k];
      Blds[kk * 64 + n] = (_Float16)v;
    }
    __syncthreads();

    // ---- build fragments: one A frag, four B frags, 4 WMMAs ----
    v16h a;
    const int ar  = (wave << 4) + (lane & 15);
    const int k0l = (lane < 16) ? 0 : 8;
    #pragma unroll
    for (int j = 0; j < 8; ++j) {
      a[j]     = Alds[ar * 32 + k0l + j];
      a[j + 8] = Alds[ar * 32 + k0l + 16 + j];
    }
    #pragma unroll
    for (int nf = 0; nf < 4; ++nf) {
      v16h b;
      #pragma unroll
      for (int j = 0; j < 16; ++j) b[j] = Blds[(k0l + (j & 7) + ((j >> 3) << 4)) * 64 + (nf << 4) + (lane & 15)];
      acc[nf] = WMMA_F16(a, b, acc[nf]);
    }
    __syncthreads();
  }

  // ---- epilogue: bias + residual + ReLU, scatter to NCHW ----
  #pragma unroll
  for (int nf = 0; nf < 4; ++nf) {
    const int ncol = n0 + (nf << 4) + (lane & 15);
    #pragma unroll
    for (int v = 0; v < 8; ++v) {
      const int mrow = m0 + (wave << 4) + v + ((lane >> 4) << 3);
      if (mrow < Mtot && ncol < OC) {
        const int bb = mrow / (OH * OW);
        const int r  = mrow % (OH * OW);
        const int oy = r / OW, ox = r % OW;
        const long long idx = ((long long)(bb * OC + ncol) * OH + oy) * OW + ox;
        float val = acc[nf][v] + bias[ncol];
        if (res) val += res[idx];
        if (relu_out) val = val > 0.f ? val : 0.f;
        out[idx] = val;
      }
    }
  }
}

// ---------------------------------------------------------------------------
// NCHW (B=16, C=64, H=64, W=64) -> rows [B*H*W, 64]
// ---------------------------------------------------------------------------
__global__ void nchw_to_rows(const float* __restrict__ ze, float* __restrict__ rows) {
  const int i = blockIdx.x * 256 + threadIdx.x;
  if (i >= 16 * 64 * 64 * 64) return;
  const int d = i & 63;
  const int m = i >> 6;
  const int b = m >> 12;
  const int r = m & 4095;
  const int y = r >> 6, x = r & 63;
  rows[i] = ze[((b * 64 + d) * 64 + y) * 64 + x];
}

// rows [B*H*W, 64] -> NCHW (i is the NCHW flat index)
__global__ void rows_to_nchw(const float* __restrict__ rows, float* __restrict__ zq) {
  const int i = blockIdx.x * 256 + threadIdx.x;
  if (i >= 16 * 64 * 64 * 64) return;
  const int x = i & 63;
  const int y = (i >> 6) & 63;
  const int d = (i >> 12) & 63;
  const int b = i >> 18;
  zq[i] = rows[(((b << 12) + (y << 6) + x) << 6) + d];
}

// ---------------------------------------------------------------------------
// Zero q_sum / sse / counts
// ---------------------------------------------------------------------------
__global__ void vq_init(float* __restrict__ qsum, float* __restrict__ sse,
                        float* __restrict__ counts) {
  const int i = blockIdx.x * 256 + threadIdx.x;
  if (i < 16 * 64 * 64 * 64) qsum[i] = 0.f;
  if (i < 4 * 512) counts[i] = 0.f;
  if (i < 4) sse[i] = 0.f;
}

// ---------------------------------------------------------------------------
// One residual-VQ level. 128 threads = 4 waves, 64 rows per block
// (16 rows per wave; cnorm + codebook B-tiles shared across waves).
// Distances d[m,n] = |r_m|^2 + |c_n|^2 - 2 <r_m, c_n>, dot via 2x f16 WMMA (K=64).
// Then argmin over 512 codes, histogram, SSE, q_sum +=, residual -= (in place).
// ---------------------------------------------------------------------------
__global__ __launch_bounds__(128) void vq_level(
    float* __restrict__ residual, float* __restrict__ qsum,
    const float* __restrict__ cb, float* __restrict__ sse,
    float* __restrict__ counts)
{
  __shared__ float    rcache[64 * 64];   // 16 KB
  __shared__ _Float16 rh[64 * 64];       // A tiles [m][k], 8 KB
  __shared__ _Float16 chT[64 * 16];      // shared B tile [k][n], 2 KB
  __shared__ float    cnorm[512];
  __shared__ float    rnorm[64];
  __shared__ float    dmat[4 * 256];     // wave-private 16x16 distance tiles
  __shared__ float    bestd[64];
  __shared__ int      bestc[64];

  const int t    = threadIdx.x;
  const int wave = t >> 5;
  const int lane = t & 31;
  const int m0   = blockIdx.x * 64;

  // load 64 residual rows (coalesced), keep fp32 + f16 copies
  #pragma unroll
  for (int i = 0; i < 32; ++i) {
    const int e = t + (i << 7);
    const float v = residual[(long long)m0 * 64 + e];
    rcache[e] = v;
    rh[e] = (_Float16)v;
  }
  // codebook norms: 512 codes, 4 per thread (L2-resident codebook)
  #pragma unroll
  for (int c = t * 4; c < t * 4 + 4; ++c) {
    float s = 0.f;
    #pragma unroll 8
    for (int d = 0; d < 64; ++d) { const float v = cb[c * 64 + d]; s += v * v; }
    cnorm[c] = s;
  }
  __syncthreads();
  if (t < 64) {
    float s = 0.f;
    #pragma unroll 8
    for (int d = 0; d < 64; ++d) { const float v = rcache[t * 64 + d]; s += v * v; }
    rnorm[t] = s;
    bestd[t] = 3.4e38f;
    bestc[t] = 0;
  }
  __syncthreads();

  const int k0l = (lane < 16) ? 0 : 8;
  for (int ct = 0; ct < 32; ++ct) {
    const int n0 = ct << 4;
    // stage 16 codes as shared B tile [k][n]
    #pragma unroll
    for (int i = 0; i < 8; ++i) {
      const int e = t + (i << 7);
      const int k = e >> 4, n = e & 15;
      chT[e] = (_Float16)cb[(n0 + n) * 64 + k];
    }
    __syncthreads();

    v8f acc = {};
    #pragma unroll
    for (int s = 0; s < 2; ++s) {
      v16h a, b;
      const int kb = s * 32;
      const int ar = (wave << 4) + (lane & 15);
      #pragma unroll
      for (int j = 0; j < 8; ++j) {
        a[j]     = rh[ar * 64 + kb + k0l + j];
        a[j + 8] = rh[ar * 64 + kb + k0l + 16 + j];
      }
      #pragma unroll
      for (int j = 0; j < 16; ++j) b[j] = chT[(kb + k0l + (j & 7) + ((j >> 3) << 4)) * 16 + (lane & 15)];
      acc = WMMA_F16(a, b, acc);
    }
    // distances (wave-private tile): VGPR v -> row = v + 8*(lane>=16), col = lane&15
    #pragma unroll
    for (int v = 0; v < 8; ++v) {
      const int row = v + ((lane >> 4) << 3);
      const int n   = lane & 15;
      dmat[(wave << 8) + row * 16 + n] = rnorm[(wave << 4) + row] + cnorm[n0 + n] - 2.f * acc[v];
    }
    // wave-local argmin scan (same-wave LDS ordering; no barrier needed)
    if (lane < 16) {
      const int gr = (wave << 4) + lane;
      float bd = bestd[gr];
      int   bc = bestc[gr];
      #pragma unroll
      for (int n = 0; n < 16; ++n) {
        const float d = dmat[(wave << 8) + lane * 16 + n];
        if (d < bd) { bd = d; bc = n0 + n; }   // strict < keeps first-min (argmin semantics)
      }
      bestd[gr] = bd; bestc[gr] = bc;
    }
    __syncthreads();
  }

  if (lane < 16) atomicAdd(&counts[bestc[(wave << 4) + lane]], 1.0f);

  // q gather, q_sum accumulate, residual update (in place), SSE reduce
  float errsq = 0.f;
  #pragma unroll
  for (int i = 0; i < 32; ++i) {
    const int e   = t + (i << 7);
    const int row = e >> 6, d = e & 63;
    const float q  = cb[bestc[row] * 64 + d];
    const float r  = rcache[e];
    const float df = q - r;
    errsq += df * df;
    const long long gi = (long long)(m0 + row) * 64 + d;
    qsum[gi]    += q;       // straight-through forward value is q
    residual[gi] = r - q;
  }
  #pragma unroll
  for (int off = 16; off > 0; off >>= 1) errsq += __shfl_xor(errsq, off, 32);
  if (lane == 0) atomicAdd(sse, errsq);
}

// ---------------------------------------------------------------------------
// rq_loss + perplexities
// ---------------------------------------------------------------------------
__global__ __launch_bounds__(512) void vq_finalize(
    const float* __restrict__ sse, const float* __restrict__ counts,
    float* __restrict__ rq_out, float* __restrict__ perp_out)
{
  __shared__ float red[512];
  const int t = threadIdx.x;
  const float invN = 1.f / 65536.f;
  for (int lvl = 0; lvl < 4; ++lvl) {
    const float p = counts[lvl * 512 + t] * invN;
    red[t] = p * logf(p + 1e-10f);
    __syncthreads();
    for (int s = 256; s > 0; s >>= 1) {
      if (t < s) red[t] += red[t + s];
      __syncthreads();
    }
    if (t == 0) perp_out[lvl] = expf(-red[0]);
    __syncthreads();
  }
  if (t == 0) {
    const float s = sse[0] + sse[1] + sse[2] + sse[3];
    rq_out[0] = 0.25f * s / (4.0f * 65536.f * 64.f);   // mean over levels of commitment loss
  }
}

// ---------------------------------------------------------------------------
// Host-side orchestration
// ---------------------------------------------------------------------------
static void conv_call(hipStream_t s, const float* in, const float* w, const float* bias,
                      const float* res, float* out,
                      int B, int IC, int IH, int IW, int OC, int OH, int OW,
                      int KH, int KW, int stride, int pad, int ldil,
                      int relu_in, int relu_out)
{
  const int M = B * OH * OW;
  dim3 grid((M + 127) / 128, (OC + 63) / 64);
  conv_wmma<<<grid, 256, 0, s>>>(in, w, bias, res, out, B, IC, IH, IW, OC, OH, OW,
                                 KH, KW, stride, pad, ldil, relu_in, relu_out);
}

extern "C" void kernel_launch(void* const* d_in, const int* in_sizes, int n_in,
                              void* d_out, int out_size, void* d_ws, size_t ws_size,
                              hipStream_t stream) {
  (void)in_sizes; (void)n_in; (void)out_size; (void)ws_size;

  const float* x       = (const float*)d_in[0];
  const float* enc_w1  = (const float*)d_in[1];
  const float* enc_b1  = (const float*)d_in[2];
  const float* enc_w2  = (const float*)d_in[3];
  const float* enc_b2  = (const float*)d_in[4];
  const float* enc_w3  = (const float*)d_in[5];
  const float* enc_b3  = (const float*)d_in[6];
  const float* enc_rw1 = (const float*)d_in[7];    // [2,32,256,3,3]
  const float* enc_rb1 = (const float*)d_in[8];    // [2,32]
  const float* enc_rw2 = (const float*)d_in[9];    // [2,256,32,1,1]
  const float* enc_rb2 = (const float*)d_in[10];   // [2,256]
  const float* enc_wp  = (const float*)d_in[11];
  const float* enc_bp  = (const float*)d_in[12];
  const float* dec_w1  = (const float*)d_in[13];
  const float* dec_b1  = (const float*)d_in[14];
  const float* dec_rw1 = (const float*)d_in[15];
  const float* dec_rb1 = (const float*)d_in[16];
  const float* dec_rw2 = (const float*)d_in[17];
  const float* dec_rb2 = (const float*)d_in[18];
  const float* dec_t1w = (const float*)d_in[19];
  const float* dec_t1b = (const float*)d_in[20];
  const float* dec_t2w = (const float*)d_in[21];
  const float* dec_t2b = (const float*)d_in[22];
  const float* cbooks  = (const float*)d_in[23];   // [4,512,64]

  // Output layout (flat, return order): x_recon, rq_loss, z_q, z_e, perps
  float* out   = (float*)d_out;
  float* xrec  = out;                              // 16*3*256*256 = 3,145,728
  float* rq    = out + 3145728;                    // 1
  float* zq    = out + 3145729;                    // 16*64*64*64 = 4,194,304
  float* ze    = out + 3145729 + 4194304;          // 4,194,304
  float* perps = out + 3145729 + 2 * 4194304;      // 4

  // Workspace layout (float elements)
  float* ws     = (float*)d_ws;
  float* buf1   = ws;                         // 16*128*128*128 = 33,554,432
  float* buf2   = ws + 33554432;              // 16*256*64*64   = 16,777,216
  float* buf3   = ws + 50331648;              // 16*256*64*64
  float* buf4   = ws + 67108864;              // 16*32*64*64    =  2,097,152
  float* buf5   = ws + 69206016;              // 16*256*64*64
  float* resid  = ws + 85983232;              // 65536*64       =  4,194,304
  float* qsum   = ws + 90177536;              // 65536*64
  float* sse    = ws + 94371840;              // 4
  float* counts = ws + 94371844;              // 4*512

  // ---------------- Encoder ----------------
  conv_call(stream, x,    enc_w1, enc_b1, nullptr, buf1, 16,   3, 256, 256, 128, 128, 128, 4, 4, 2, 1, 1, 0, 1);
  conv_call(stream, buf1, enc_w2, enc_b2, nullptr, buf2, 16, 128, 128, 128, 256,  64,  64, 4, 4, 2, 1, 1, 0, 1);
  conv_call(stream, buf2, enc_w3, enc_b3, nullptr, buf3, 16, 256,  64,  64, 256,  64,  64, 3, 3, 1, 1, 1, 0, 0);
  // res block 0: h = conv1x1(relu(conv3x3(relu(x)))); x += h
  conv_call(stream, buf3, enc_rw1,              enc_rb1,       nullptr, buf4, 16, 256, 64, 64,  32, 64, 64, 3, 3, 1, 1, 1, 1, 0);
  conv_call(stream, buf4, enc_rw2,              enc_rb2,       buf3,    buf5, 16,  32, 64, 64, 256, 64, 64, 1, 1, 1, 0, 1, 1, 0);
  // res block 1 (+ final stack ReLU)
  conv_call(stream, buf5, enc_rw1 + 32*256*9,   enc_rb1 + 32,  nullptr, buf4, 16, 256, 64, 64,  32, 64, 64, 3, 3, 1, 1, 1, 1, 0);
  conv_call(stream, buf4, enc_rw2 + 256*32,     enc_rb2 + 256, buf5,    buf3, 16,  32, 64, 64, 256, 64, 64, 1, 1, 1, 0, 1, 1, 1);
  // projection -> z_e (written straight into d_out)
  conv_call(stream, buf3, enc_wp, enc_bp, nullptr, ze, 16, 256, 64, 64, 64, 64, 64, 1, 1, 1, 0, 1, 0, 0);

  // ---------------- Residual VQ ----------------
  const int NE = 16 * 64 * 64 * 64;
  vq_init<<<(NE + 255) / 256, 256, 0, stream>>>(qsum, sse, counts);
  nchw_to_rows<<<(NE + 255) / 256, 256, 0, stream>>>(ze, resid);
  for (int lvl = 0; lvl < 4; ++lvl) {
    vq_level<<<65536 / 64, 128, 0, stream>>>(resid, qsum, cbooks + (long long)lvl * 512 * 64,
                                             sse + lvl, counts + lvl * 512);
  }
  vq_finalize<<<1, 512, 0, stream>>>(sse, counts, rq, perps);
  rows_to_nchw<<<(NE + 255) / 256, 256, 0, stream>>>(qsum, zq);

  // ---------------- Decoder ----------------
  conv_call(stream, zq,   dec_w1, dec_b1, nullptr, buf2, 16, 64, 64, 64, 256, 64, 64, 3, 3, 1, 1, 1, 0, 0);
  conv_call(stream, buf2, dec_rw1,             dec_rb1,       nullptr, buf4, 16, 256, 64, 64,  32, 64, 64, 3, 3, 1, 1, 1, 1, 0);
  conv_call(stream, buf4, dec_rw2,             dec_rb2,       buf2,    buf5, 16,  32, 64, 64, 256, 64, 64, 1, 1, 1, 0, 1, 1, 0);
  conv_call(stream, buf5, dec_rw1 + 32*256*9,  dec_rb1 + 32,  nullptr, buf4, 16, 256, 64, 64,  32, 64, 64, 3, 3, 1, 1, 1, 1, 0);
  conv_call(stream, buf4, dec_rw2 + 256*32,    dec_rb2 + 256, buf5,    buf3, 16,  32, 64, 64, 256, 64, 64, 1, 1, 1, 0, 1, 1, 1);
  // ConvTranspose2d(k=4, s=2, p=1) == conv with lhs_dilation=2, pad=2, stride=1
  conv_call(stream, buf3, dec_t1w, dec_t1b, nullptr, buf1, 16, 256,  64,  64, 128, 128, 128, 4, 4, 1, 2, 2, 0, 1);
  conv_call(stream, buf1, dec_t2w, dec_t2b, nullptr, xrec, 16, 128, 128, 128,   3, 256, 256, 4, 4, 1, 2, 2, 0, 0);
}
